// EncodecResidualVectorQuantizer_23356032156249
// MI455X (gfx1250) — compile-verified, matrix operand order checked by reading, not run on hardware
//
#include <hip/hip_runtime.h>
#include <hip/hip_bf16.h>

typedef __attribute__((ext_vector_type(16))) __bf16 v16bf;
typedef __attribute__((ext_vector_type(8)))  float  v8f;

#define B_  32
#define D_  128
#define T_  2250
#define Q_  32
#define K_  1024
#define N_  (B_ * T_)            /* 72000 rows */

/* workspace layout (bytes) */
#define R_BYTES   ((size_t)N_ * D_ * 4)          /* residual fp32: 36,864,000 */
#define EHI_BYTES ((size_t)Q_ * K_ * D_ * 2)     /* swizzled bf16 hi: 8,388,608 */
#define ELO_OFF   (R_BYTES + EHI_BYTES)
#define CN_OFF    (R_BYTES + 2 * EHI_BYTES)      /* ||e||^2 fp32: Q*K floats */

/* ---------------- prep 1: [B,D,T] -> r[N=B*T, D] fp32 (LDS tiled transpose) -------- */
__global__ void rvq_transpose_kernel(const float* __restrict__ emb, float* __restrict__ r) {
    __shared__ float tile[32][33];
    const int tt = blockIdx.x, dt = blockIdx.y, b = blockIdx.z;
    const int tx = threadIdx.x, ty = threadIdx.y;   /* block 32x8 */
    const int t0 = tt * 32, d0 = dt * 32;
#pragma unroll
    for (int i = 0; i < 4; ++i) {
        int d = d0 + ty + i * 8;
        int t = t0 + tx;
        float v = 0.0f;
        if (t < T_) v = emb[((size_t)b * D_ + d) * T_ + t];
        tile[ty + i * 8][tx] = v;
    }
    __syncthreads();
#pragma unroll
    for (int i = 0; i < 4; ++i) {
        int t = t0 + ty + i * 8;
        int d = d0 + tx;
        if (t < T_) r[((size_t)b * T_ + t) * D_ + d] = tile[tx][ty + i * 8];
    }
}

/* ------- prep 2: code norms + bf16 hi/lo codebook pre-swizzled into WMMA-B order ---- */
__global__ void rvq_prep_codebook(const float* __restrict__ cb,
                                  __bf16* __restrict__ ehi, __bf16* __restrict__ elo,
                                  float* __restrict__ cn) {
    int id = blockIdx.x * blockDim.x + threadIdx.x;     /* one thread per (q,k) */
    if (id >= Q_ * K_) return;
    const int q = id / K_, k = id % K_;
    const int g = k >> 4, col = k & 15;
    const float* e = cb + (size_t)id * D_;
    float norm = 0.0f;
    for (int d = 0; d < D_; ++d) {
        float x = e[d];
        norm += x * x;
        __bf16 hi = (__bf16)x;
        __bf16 lo = (__bf16)(x - (float)hi);
        int c = d >> 5, dd = d & 31;
        int h, j;
        if (dd < 16) { h = dd >> 3; j = dd & 7; }
        else         { int e2 = dd - 16; h = e2 >> 3; j = 8 + (e2 & 7); }
        size_t off = (((size_t)(q * 64 + g)) * 4 + c) * 512 + (size_t)(col + 16 * h) * 16 + j;
        ehi[off] = hi;
        elo[off] = lo;
    }
    cn[id] = norm;
}

/* -------- fused per-layer kernel: async-LDS B staging + WMMA + argmax + update ------ */
__global__ __launch_bounds__(256)
void rvq_layer_kernel(float* __restrict__ r,
                      const __bf16* __restrict__ ehi, const __bf16* __restrict__ elo,
                      const float* __restrict__ cn, const float* __restrict__ cb,
                      float* __restrict__ out, int q) {
    /* double-buffered code group: [hi 4KB | lo 4KB] per buffer */
    __shared__ __bf16 sbuf[2][4096];

    const int tid  = threadIdx.x;
    const int lane = tid & 31;
    const int w    = tid >> 5;                 /* 8 waves x 32 rows = 256 rows/block */
    const int rowBase = blockIdx.x * 256 + w * 32;
    const int h = lane >> 4;

    /* A operands: 2 M-tiles of 16 rows; fp32 -> bf16 hi + lo split in registers. */
    v16bf ahi[2][4], alo[2][4];
#pragma unroll
    for (int t = 0; t < 2; ++t) {
        int row = rowBase + t * 16 + (lane & 15);
        if (row >= N_) row = N_ - 1;
#pragma unroll
        for (int c = 0; c < 4; ++c) {
            const float* base = r + (size_t)row * D_ + c * 32 + h * 8;
#pragma unroll
            for (int half = 0; half < 2; ++half) {
                const float4* p = (const float4*)(base + half * 16);
                float4 f0 = p[0], f1 = p[1];
                float v[8] = { f0.x, f0.y, f0.z, f0.w, f1.x, f1.y, f1.z, f1.w };
#pragma unroll
                for (int j = 0; j < 8; ++j) {
                    __bf16 hi = (__bf16)v[j];
                    ahi[t][c][half * 8 + j] = hi;
                    alo[t][c][half * 8 + j] = (__bf16)(v[j] - (float)hi);
                }
            }
        }
    }

    float best[2][8];
    int   bidx[2][8];
#pragma unroll
    for (int t = 0; t < 2; ++t) {
#pragma unroll
        for (int v = 0; v < 8; ++v) { best[t][v] = -3.4e38f; bidx[t][v] = 0; }
    }

    const float* cnq = cn + (size_t)q * K_;
    const char* ghi = (const char*)ehi + (size_t)q * 64 * 4096;   /* 4KB per group */
    const char* glo = (const char*)elo + (size_t)q * 64 * 4096;
    const unsigned sbase = (unsigned)(unsigned long long)(void*)&sbuf[0][0];

    /* cooperative async stage of group g into buffer b: 256 thr x 16B x {hi,lo} = 8KB */
    auto stage = [&](int g, int b) {
        unsigned long long ah = (unsigned long long)(ghi + (size_t)g * 4096 + tid * 16);
        unsigned long long al = (unsigned long long)(glo + (size_t)g * 4096 + tid * 16);
        unsigned dh = sbase + (unsigned)b * 8192u + (unsigned)tid * 16u;
        unsigned dl = dh + 4096u;
        asm volatile("global_load_async_to_lds_b128 %0, %1, off" :: "v"(dh), "v"(ah) : "memory");
        asm volatile("global_load_async_to_lds_b128 %0, %1, off" :: "v"(dl), "v"(al) : "memory");
    };

    stage(0, 0);
    for (int g = 0; g < 64; ++g) {                   /* 64 groups of 16 codes */
        asm volatile("s_wait_asynccnt 0x0" ::: "memory");
        __syncthreads();                              /* buffer g&1 ready for all waves */
        if (g + 1 < 64) stage(g + 1, (g + 1) & 1);    /* overlap next group's DMA */

        const __bf16* sh = &sbuf[g & 1][0];
        v8f acc0 = {}, acc1 = {};
#pragma unroll
        for (int c = 0; c < 4; ++c) {                 /* D = 4 chunks of 32 */
            v16bf bh = *(const v16bf*)(sh + c * 512 + lane * 16);
            v16bf bl = *(const v16bf*)(sh + 2048 + c * 512 + lane * 16);
            /* two independent accumulation chains, 3-term split (hi*hi + hi*lo + lo*hi) per tile */
            acc0 = __builtin_amdgcn_wmma_f32_16x16x32_bf16(false, ahi[0][c], false, bh, (short)0, acc0, false, false);
            acc1 = __builtin_amdgcn_wmma_f32_16x16x32_bf16(false, ahi[1][c], false, bh, (short)0, acc1, false, false);
            acc0 = __builtin_amdgcn_wmma_f32_16x16x32_bf16(false, ahi[0][c], false, bl, (short)0, acc0, false, false);
            acc1 = __builtin_amdgcn_wmma_f32_16x16x32_bf16(false, ahi[1][c], false, bl, (short)0, acc1, false, false);
            acc0 = __builtin_amdgcn_wmma_f32_16x16x32_bf16(false, alo[0][c], false, bh, (short)0, acc0, false, false);
            acc1 = __builtin_amdgcn_wmma_f32_16x16x32_bf16(false, alo[1][c], false, bh, (short)0, acc1, false, false);
        }
        const int   kcode = g * 16 + (lane & 15);
        const float c2    = cnq[kcode];
#pragma unroll
        for (int v = 0; v < 8; ++v) {
            float s0 = 2.0f * acc0[v] - c2;           /* ||x||^2 row-constant: dropped */
            float s1 = 2.0f * acc1[v] - c2;
            if (s0 > best[0][v]) { best[0][v] = s0; bidx[0][v] = kcode; }
            if (s1 > best[1][v]) { best[1][v] = s1; bidx[1][v] = kcode; }
        }
        __syncthreads();                              /* done reading buf g&1 before re-stage */
    }

    /* argmax across the 16 lanes sharing each row (prefer smaller index on tie) */
#pragma unroll
    for (int off = 8; off >= 1; off >>= 1) {
#pragma unroll
        for (int t = 0; t < 2; ++t) {
#pragma unroll
            for (int v = 0; v < 8; ++v) {
                float os = __shfl_xor(best[t][v], off, 32);
                int   oi = __shfl_xor(bidx[t][v], off, 32);
                if (os > best[t][v] || (os == best[t][v] && oi < bidx[t][v])) {
                    best[t][v] = os; bidx[t][v] = oi;
                }
            }
        }
    }

    /* emit indices and update residual r -= e[idx] in exact fp32 */
    const float* cbq = cb + (size_t)q * K_ * D_;
#pragma unroll
    for (int t = 0; t < 2; ++t) {
#pragma unroll
        for (int m = 0; m < 16; ++m) {
            int code = __shfl(bidx[t][m & 7], (m < 8) ? 0 : 16, 32);
            int row  = rowBase + t * 16 + m;
            if (row < N_) {
                if (lane == 0) out[(size_t)q * N_ + row] = (float)code;
                float4*       rv = (float4*)(r + (size_t)row * D_);
                const float4* ev = (const float4*)(cbq + (size_t)code * D_);
                float4 a = rv[lane];
                float4 e = ev[lane];
                a.x -= e.x; a.y -= e.y; a.z -= e.z; a.w -= e.w;
                rv[lane] = a;
            }
        }
    }
}

extern "C" void kernel_launch(void* const* d_in, const int* in_sizes, int n_in,
                              void* d_out, int out_size, void* d_ws, size_t ws_size,
                              hipStream_t stream) {
    (void)in_sizes; (void)n_in; (void)out_size; (void)ws_size;
    const float* emb = (const float*)d_in[0];     /* [B, D, T] fp32 */
    const float* cb  = (const float*)d_in[1];     /* [Q, K, D] fp32 */
    float* out = (float*)d_out;                   /* [Q, B, T] indices (as fp32 values) */

    char*   ws  = (char*)d_ws;
    float*  r   = (float*)ws;                     /* residual [N, D] fp32 */
    __bf16* ehi = (__bf16*)(ws + R_BYTES);        /* swizzled bf16 hi codebook */
    __bf16* elo = (__bf16*)(ws + ELO_OFF);        /* swizzled bf16 lo codebook */
    float*  cn  = (float*)(ws + CN_OFF);          /* ||e||^2 per (q,k) */

    /* 1) x = transpose(embeddings).reshape(N, D) */
    rvq_transpose_kernel<<<dim3((T_ + 31) / 32, D_ / 32, B_), dim3(32, 8), 0, stream>>>(emb, r);
    /* 2) codebook norms + WMMA-swizzled bf16 hi/lo tables */
    rvq_prep_codebook<<<(Q_ * K_ + 255) / 256, 256, 0, stream>>>(cb, ehi, elo, cn);
    /* 3) Q sequential residual-quantizer layers (stream-ordered) */
    const int blocks = (N_ + 255) / 256;
    for (int q = 0; q < Q_; ++q) {
        rvq_layer_kernel<<<blocks, 256, 0, stream>>>(r, ehi, elo, cn, cb, out, q);
    }
}
